// FRKTNet_60043642798119
// MI455X (gfx1250) — matrix operand order repaired and checked by reading, not compiled, not run on hardware
//
#include <hip/hip_runtime.h>

// FRKT-style knowledge-tracing net, CDNA5 (gfx1250, wave32, WMMA).
// v3: loop-invariant WMMA weights (gru_w_hh, Wf[:, :128]) staged in LDS so the
// recurrent-step B-tiles are ds_load_b128 instead of scratch spill reloads.
#define BB  64
#define TT  500
#define HH  128
#define KK  32
#define SK1 101
#define BT  (BB * TT)

typedef __attribute__((ext_vector_type(16))) _Float16 v16h;
typedef __attribute__((ext_vector_type(8)))  _Float16 v8h;
typedef __attribute__((ext_vector_type(8)))  float    v8f;

__device__ __forceinline__ float sigmoidf_(float x) { return 1.0f / (1.0f + __expf(-x)); }

// ---- WMMA tile loaders per CDNA5 ISA layouts (wave32), f16 sources ----
// A: 16x32 f16. lanes 0-15: M=lane, K in {0..7,16..23}; lanes 16-31: M=lane-16, K in {8..15,24..31}.
__device__ __forceinline__ v16h load_a_tile_f16(const _Float16* Arow0, int lda, int k0) {
  const int lane = threadIdx.x & 31;
  const int m  = lane & 15;
  const int kh = lane >> 4;
  const _Float16* r = Arow0 + (size_t)m * lda + k0 + kh * 8;
  const v8h lo = *(const v8h*)r;          // K = kh*8 .. +7
  const v8h hi = *(const v8h*)(r + 16);   // K = 16 + kh*8 .. +7
  return __builtin_shufflevector(lo, hi, 0, 1, 2, 3, 4, 5, 6, 7,
                                 8, 9, 10, 11, 12, 13, 14, 15);
}

// B: 32x16 f16 (K x N) from row-major W16[N,ldw] (B = W^T).
// lanes 0-15: K in {0..7,16..23}; lanes 16-31: K in {8..15,24..31}; N = lane&15.
__device__ __forceinline__ v16h load_b_tile_f16(const _Float16* W, int ldw, int n0, int k0) {
  const int lane = threadIdx.x & 31;
  const int n  = n0 + (lane & 15);
  const int kh = lane >> 4;
  const _Float16* r = W + (size_t)n * ldw + k0 + kh * 8;
  const v8h lo = *(const v8h*)r;
  const v8h hi = *(const v8h*)(r + 16);
  return __builtin_shufflevector(lo, hi, 0, 1, 2, 3, 4, 5, 6, 7,
                                 8, 9, 10, 11, 12, 13, 14, 15);
}

// C/D 16x16 f32: element i -> M=(lane>>4)*8+i, N=lane&15.  Optional fused f16 copy.
__device__ __forceinline__ void store_c_tile(float* Crow0, _Float16* C16row0, int ldc, int n0,
                                             const float* bias, v8f acc, int act) {
  const int lane = threadIdx.x & 31;
  const int n  = lane & 15;
  const int mb = (lane >> 4) * 8;
  const float bv = bias ? bias[n0 + n] : 0.0f;
#pragma unroll
  for (int i = 0; i < 8; ++i) {
    float v = acc[i] + bv;
    if (act) v = fmaxf(v, 0.0f);
    Crow0[(size_t)(mb + i) * ldc + n0 + n] = v;
    if (C16row0) C16row0[(size_t)(mb + i) * ldc + n0 + n] = (_Float16)v;
  }
}

// ---------------- weight f32 -> f16 staging ----------------
__global__ void cvt_f16_kernel(const float* __restrict__ src, _Float16* __restrict__ dst, int n) {
  const int i = blockIdx.x * blockDim.x + threadIdx.x;
  if (i < n) dst[i] = (_Float16)src[i];
}

// ---------------- embedding gather + sparse q_matrix matvec ----------------
__global__ void embed_kernel(const int* __restrict__ a_data, const int* __restrict__ e_data,
                             const float* __restrict__ q_matrix, const float* __restrict__ s_embed_w,
                             const float* __restrict__ a_embed_w, const float* __restrict__ e_embed_w,
                             float* __restrict__ e_emb,
                             _Float16* __restrict__ s_emb16, _Float16* __restrict__ e_emb16,
                             _Float16* __restrict__ a_emb16) {
  const int idx = blockIdx.x * 2 + (threadIdx.x >> 7);   // (b*T + t)
  const int h = threadIdx.x & 127;
  if (idx >= BT) return;
  const int e = e_data[idx];
  const int a = a_data[idx];
  const float* qrow = q_matrix + (size_t)e * SK1;
  float sum = 0.0f, cnt = 0.0f;
  for (int s = 0; s < SK1; ++s) {
    const float r = qrow[s];
    sum += r * s_embed_w[s * HH + h];
    cnt += r;
  }
  if (cnt == 0.0f) cnt = 1.0f;
  const float ev = e_embed_w[(size_t)e * HH + h];
  s_emb16[(size_t)idx * HH + h] = (_Float16)(sum / cnt);
  e_emb[(size_t)idx * HH + h] = ev;
  e_emb16[(size_t)idx * HH + h] = (_Float16)ev;
  a_emb16[(size_t)idx * HH + h] = (_Float16)a_embed_w[(size_t)a * HH + h];
}

// --------- WMMA GEMM: C = act(cat(A0,A1,A2) @ W^T + bias) ----------
// each Ai is [M,128] f16 row-major; W is [N,Ktot] f16 row-major.
// One wave computes a 16x(16*NT) strip: 1 A tile feeds NT wmma issues.
template <int NT>
__global__ void gemm_cat_kernel(const _Float16* __restrict__ A0, const _Float16* __restrict__ A1,
                                const _Float16* __restrict__ A2, const _Float16* __restrict__ W,
                                const float* __restrict__ bias, float* __restrict__ C,
                                _Float16* __restrict__ C16, int M, int N, int Ktot, int act) {
  const int wave = (int)((blockIdx.x * (size_t)blockDim.x + threadIdx.x) >> 5);
  const int nwaves_row = (N >> 4) / NT;
  const int mt = wave / nwaves_row;
  const int n0 = (wave % nwaves_row) * NT * 16;
  if (mt * 16 >= M) return;                       // wave-uniform
  const v8f zero = {};
  v8f acc[NT];
#pragma unroll
  for (int j = 0; j < NT; ++j) acc[j] = zero;
  for (int k0 = 0; k0 < Ktot; k0 += 32) {
    const _Float16* src = (k0 < 128) ? A0 : ((k0 < 256) ? A1 : A2);
    const v16h a = load_a_tile_f16(src + (size_t)mt * 16 * HH, HH, k0 & 127);
#pragma unroll
    for (int j = 0; j < NT; ++j) {
      const v16h b = load_b_tile_f16(W, Ktot, n0 + j * 16, k0);
      acc[j] = __builtin_amdgcn_wmma_f32_16x16x32_f16(false, a, false, b, (short)0, acc[j],
                                                      false, false);
    }
  }
#pragma unroll
  for (int j = 0; j < NT; ++j)
    store_c_tile(C + (size_t)mt * 16 * N, C16 ? C16 + (size_t)mt * 16 * N : nullptr, N,
                 n0 + j * 16, bias, acc[j], act);
}

// ---------------- GRU scan: single workgroup, everything in LDS ----------------
// LDS: hS[64][128] f32 + ghS[64][384] f32 + bhh[384] f32 + hS16[64][128] f16
//      + whhS[384][128] f16  (~242KB of the WGP's 320KB).
// 16 waves x 6 tiles = 96 wmma tiles per step; both wmma operands are ds_load_b128.
__global__ void gru_kernel(const float* __restrict__ gx, const _Float16* __restrict__ whh16,
                           const float* __restrict__ bhh, const float* __restrict__ h0,
                           float* __restrict__ gru_out) {
  extern __shared__ float gsm[];
  float* hS   = gsm;                                // [64*128]
  float* ghS  = gsm + BB * HH;                      // [64*384]
  float* bhhS = ghS + BB * 384;                     // [384]
  _Float16* hS16 = (_Float16*)(bhhS + 384);         // [64*128]
  _Float16* whhS = hS16 + BB * HH;                  // [384*128]
  const int tid = threadIdx.x;
  for (int j = tid; j < BB * HH; j += 512) {
    const float v = h0[j & 127];
    hS[j] = v;
    hS16[j] = (_Float16)v;
  }
  if (tid < 384) bhhS[tid] = bhh[tid];
  {  // stage recurrent weights (f16) into LDS, 4B-chunk copy
    const unsigned* src = (const unsigned*)whh16;
    unsigned* dst = (unsigned*)whhS;
    for (int j = tid; j < (384 * HH) / 2; j += 512) dst[j] = src[j];
  }
  __syncthreads();
  const int wave = tid >> 5;
  for (int t = 0; t < TT; ++t) {
    // gh = h @ whh^T + bhh  (M=64, N=384, K=128), operands from LDS
    for (int i = 0; i < 6; ++i) {
      const int tile = wave * 6 + i;
      const int mt = tile / 24, nt = tile % 24;
      v8f acc = {};
      for (int k0 = 0; k0 < HH; k0 += 32) {
        const v16h a = load_a_tile_f16(hS16 + (size_t)mt * 16 * HH, HH, k0);
        const v16h b = load_b_tile_f16(whhS, HH, nt * 16, k0);
        acc = __builtin_amdgcn_wmma_f32_16x16x32_f16(false, a, false, b, (short)0, acc,
                                                     false, false);
      }
      store_c_tile(ghS + (size_t)mt * 16 * 384, nullptr, 384, nt * 16, bhhS, acc, 0);
    }
    __syncthreads();
    // gates + state update
    for (int j = tid; j < BB * HH; j += 512) {
      const int b = j >> 7, h = j & 127;
      const float* gxr = gx + ((size_t)b * TT + t) * 384;
      const float xr = gxr[h], xz = gxr[128 + h], xn = gxr[256 + h];
      const float hr = ghS[b * 384 + h];
      const float hz = ghS[b * 384 + 128 + h];
      const float hn = ghS[b * 384 + 256 + h];
      const float hprev = hS[j];
      const float r = sigmoidf_(xr + hr);
      const float z = sigmoidf_(xz + hz);
      const float nn = tanhf(xn + r * hn);
      const float hnew = (1.0f - z) * nn + z * hprev;
      hS[j] = hnew;
      hS16[j] = (_Float16)hnew;
      gru_out[((size_t)b * TT + t) * HH + h] = hnew;
    }
    __syncthreads();
  }
}

// ------------- key-value memory scan: one workgroup per batch --------------
// LDS (~74KB): h_pre f32+f16 shadow, gam, WfA staged as f16 [128][128].
__global__ void memscan_kernel(const float* __restrict__ Wall, const float* __restrict__ se_learn,
                               const float* __restrict__ Wg, const float* __restrict__ bg,
                               const float* __restrict__ Wgg, const float* __restrict__ bgg,
                               const float* __restrict__ Wf, const float* __restrict__ bf,
                               const float* __restrict__ m0, float* __restrict__ out1) {
  extern __shared__ float msm[];
  float* hpreS = msm;                         // [32*128]
  float* gamS  = hpreS + KK * HH;             // [32*128]
  float* htS   = gamS + KK * HH;              // [128]
  float* LG    = htS + HH;                    // [128]
  float* gateS = LG + HH;                     // [128]
  float* cvec  = gateS + HH;                  // [128]
  _Float16* hpre16 = (_Float16*)(cvec + HH);  // [32*128]
  _Float16* WfS    = hpre16 + KK * HH;        // [128*128]  (Wf[:, :128] packed ld=128)
  const int b = blockIdx.x;
  const int tid = threadIdx.x;
  // init: h_pre = m0 ; ht0 = sum_k Wall[b,0,k]*m0[k,:] ; stage WfA -> LDS f16
  for (int j = tid; j < KK * HH; j += 256) {
    const float v = m0[j];
    hpreS[j] = v;
    hpre16[j] = (_Float16)v;
  }
  for (int j = tid; j < HH * HH; j += 256)
    WfS[j] = (_Float16)Wf[(size_t)(j >> 7) * 256 + (j & 127)];
  if (tid < HH) {
    const float* w0 = Wall + (size_t)b * TT * KK;
    float s = 0.0f;
    for (int k = 0; k < KK; ++k) s += w0[k] * m0[k * HH + tid];
    htS[tid] = s;
  }
  __syncthreads();
  const int wave = tid >> 5;
  for (int t = 0; t < TT - 1; ++t) {
    const float* learn = se_learn + ((size_t)b * TT + t) * HH;
    // gain (threads 0..127 -> LG) and gate (threads 128..255 -> gateS), hl=[ht, learning]
    if (tid < HH) {
      const float* wrow = Wg + (size_t)tid * 256;
      float acc = bg[tid];
      for (int j = 0; j < HH; ++j) acc += htS[j] * wrow[j];
      for (int j = 0; j < HH; ++j) acc += learn[j] * wrow[128 + j];
      LG[tid] = tanhf(acc);
    } else {
      const int i = tid - HH;
      const float* wrow = Wgg + (size_t)i * 256;
      float acc = bgg[i];
      for (int j = 0; j < HH; ++j) acc += htS[j] * wrow[j];
      for (int j = 0; j < HH; ++j) acc += learn[j] * wrow[128 + j];
      gateS[i] = sigmoidf_(acc);
    }
    __syncthreads();
    if (tid < HH) LG[tid] *= gateS[tid];
    __syncthreads();
    // cvec[n] = bf[n] + LG . Wf[n,128:256]  (shared across all K rows)
    if (tid < HH) {
      const float* wrow = Wf + (size_t)tid * 256 + 128;
      float acc = bf[tid];
      for (int j = 0; j < HH; ++j) acc += LG[j] * wrow[j];
      cvec[tid] = acc;
    }
    __syncthreads();
    // gam = sigmoid(h_pre @ WfA^T + cvec)  : M=32, N=128, K=128 -> 16 tiles, 8 waves
    for (int i = 0; i < 2; ++i) {
      const int tile = wave * 2 + i;
      const int mt = tile >> 3, nt = tile & 7;
      v8f acc = {};
      for (int k0 = 0; k0 < HH; k0 += 32) {
        const v16h a  = load_a_tile_f16(hpre16 + (size_t)mt * 16 * HH, HH, k0);
        const v16h bm = load_b_tile_f16(WfS, HH, nt * 16, k0);
        acc = __builtin_amdgcn_wmma_f32_16x16x32_f16(false, a, false, bm, (short)0, acc,
                                                     false, false);
      }
      const int lane = threadIdx.x & 31;
      const int n = nt * 16 + (lane & 15);
      const int mb = (lane >> 4) * 8;
#pragma unroll
      for (int e = 0; e < 8; ++e)
        gamS[(size_t)(mt * 16 + mb + e) * HH + n] = sigmoidf_(acc[e] + cvec[n]);
    }
    __syncthreads();
    // h = gam*h_pre + w ⊗ LG
    const float* wv = Wall + ((size_t)b * TT + t) * KK;
    for (int j = tid; j < KK * HH; j += 256) {
      const int k = j >> 7, h = j & 127;
      const float v = gamS[j] * hpreS[j] + wv[k] * LG[h];
      hpreS[j] = v;
      hpre16[j] = (_Float16)v;
    }
    __syncthreads();
    // ht = sum_k w_next[k] * h[k,:]
    if (tid < HH) {
      const float* wn = Wall + ((size_t)b * TT + t + 1) * KK;
      float s = 0.0f;
      for (int k = 0; k < KK; ++k) s += wn[k] * hpreS[k * HH + tid];
      htS[tid] = s;
      out1[((size_t)b * (TT - 1) + t) * HH + tid] = s;
    }
    __syncthreads();
  }
}

// ---------------- output heads: two 384-dots + sigmoid product -------------
__global__ void output_kernel(const float* __restrict__ gru_out, const float* __restrict__ out1,
                              const float* __restrict__ e_emb, const float* __restrict__ se_data,
                              const float* __restrict__ Wp, const float* __restrict__ bp,
                              const float* __restrict__ Wp1, const float* __restrict__ bp1,
                              float* __restrict__ pred) {
  const int idx = blockIdx.x * blockDim.x + threadIdx.x;    // over B*(T-1)
  if (idx >= BB * (TT - 1)) return;
  const int b = idx / (TT - 1), t = idx % (TT - 1);
  const float* o  = gru_out + ((size_t)b * TT + t) * HH;        // out[:, t]
  const float* o1 = out1 + (size_t)idx * HH;                    // out1[:, t]
  const float* ee = e_emb + ((size_t)b * TT + t + 1) * HH;      // e_emb[:, t+1]
  const float* sd = se_data + ((size_t)b * TT + t + 1) * HH;    // se_data[:, t+1]
  float acc = bp[0], acc1 = bp1[0];
  for (int j = 0; j < HH; ++j) {
    acc  += o[j] * Wp[j]  + o1[j] * Wp[128 + j]  + ee[j] * Wp[256 + j];
    acc1 += o[j] * Wp1[j] + o1[j] * Wp1[128 + j] + sd[j] * Wp1[256 + j];
  }
  pred[(size_t)b * TT + t + 1] = sigmoidf_(acc) * sigmoidf_(acc1);
  if (t == 0) pred[(size_t)b * TT] = 0.0f;
}

extern "C" void kernel_launch(void* const* d_in, const int* in_sizes, int n_in,
                              void* d_out, int out_size, void* d_ws, size_t ws_size,
                              hipStream_t stream) {
  (void)in_sizes; (void)n_in; (void)out_size; (void)ws_size;
  const int*   a_data    = (const int*)d_in[1];
  const int*   e_data    = (const int*)d_in[2];
  const float* q_matrix  = (const float*)d_in[4];
  const float* s_embed_w = (const float*)d_in[5];
  const float* a_embed_w = (const float*)d_in[6];
  const float* e_embed_w = (const float*)d_in[7];
  const float* W1 = (const float*)d_in[8];   const float* b1 = (const float*)d_in[9];
  const float* W2 = (const float*)d_in[10];  const float* b2 = (const float*)d_in[11];
  const float* W3 = (const float*)d_in[12];  const float* b3 = (const float*)d_in[13];
  const float* key_matrix = (const float*)d_in[14];
  const float* gw_ih = (const float*)d_in[15];
  const float* gw_hh = (const float*)d_in[16];
  const float* gb_ih = (const float*)d_in[17];
  const float* gb_hh = (const float*)d_in[18];
  const float* Wg  = (const float*)d_in[19]; const float* bg  = (const float*)d_in[20];
  const float* Wgg = (const float*)d_in[21]; const float* bgg = (const float*)d_in[22];
  const float* Wf  = (const float*)d_in[23]; const float* bf  = (const float*)d_in[24];
  const float* Wp  = (const float*)d_in[25]; const float* bp  = (const float*)d_in[26];
  const float* Wp1 = (const float*)d_in[27]; const float* bp1 = (const float*)d_in[28];
  const float* h0  = (const float*)d_in[29];
  const float* m0  = (const float*)d_in[30];
  float* pred = (float*)d_out;

  // ---- workspace carve: f32 region first, then f16 region (16B aligned) ----
  float* ws = (float*)d_ws;
  float* e_emb    = ws;
  float* se_data  = e_emb    + (size_t)BT * HH;
  float* se_learn = se_data  + (size_t)BT * HH;
  float* e_learn  = se_learn + (size_t)BT * HH;
  float* gru_out  = e_learn  + (size_t)BT * HH;
  float* gx       = gru_out  + (size_t)BT * HH;
  float* WallB    = gx       + (size_t)BT * 384;
  float* out1     = WallB    + (size_t)BT * KK;
  _Float16* h16   = (_Float16*)(out1 + (size_t)BB * (TT - 1) * HH);
  _Float16* s_emb16   = h16;
  _Float16* e_emb16   = s_emb16   + (size_t)BT * HH;
  _Float16* a_emb16   = e_emb16   + (size_t)BT * HH;
  _Float16* se_data16 = a_emb16   + (size_t)BT * HH;
  _Float16* e_learn16 = se_data16 + (size_t)BT * HH;
  _Float16* W1_16     = e_learn16 + (size_t)BT * HH;
  _Float16* W2_16     = W1_16  + 128 * 256;
  _Float16* W3_16     = W2_16  + 128 * 384;
  _Float16* gwih16    = W3_16  + 128 * 256;
  _Float16* gwhh16    = gwih16 + 384 * 128;
  _Float16* key16     = gwhh16 + 384 * 128;

  // ---- stage weights to f16 ----
  auto cvt = [&](const float* s, _Float16* d, int n) {
    cvt_f16_kernel<<<(n + 255) / 256, 256, 0, stream>>>(s, d, n);
  };
  cvt(W1, W1_16, 128 * 256);
  cvt(W2, W2_16, 128 * 384);
  cvt(W3, W3_16, 128 * 256);
  cvt(gw_ih, gwih16, 384 * 128);
  cvt(gw_hh, gwhh16, 384 * 128);
  cvt(key_matrix, key16, 32 * 128);

  embed_kernel<<<BT / 2, 256, 0, stream>>>(a_data, e_data, q_matrix, s_embed_w,
                                           a_embed_w, e_embed_w, e_emb,
                                           s_emb16, e_emb16, a_emb16);

  auto blocks_for = [&](int N, int NT) {
    const size_t waves = (size_t)(BT / 16) * ((N / 16) / NT);
    return (int)((waves * 32 + 255) / 256);
  };
  // se_data = relu([s,e] @ W1^T + b1), fused f16 copy for Wall GEMM
  gemm_cat_kernel<4><<<blocks_for(128, 4), 256, 0, stream>>>(
      s_emb16, e_emb16, nullptr, W1_16, b1, se_data, se_data16, BT, 128, 256, 1);
  // se_learning = relu([s,e,a] @ W2^T + b2)
  gemm_cat_kernel<4><<<blocks_for(128, 4), 256, 0, stream>>>(
      s_emb16, e_emb16, a_emb16, W2_16, b2, se_learn, nullptr, BT, 128, 384, 1);
  // e_learning = relu([e,a] @ W3^T + b3), fused f16 copy for gx GEMM
  gemm_cat_kernel<4><<<blocks_for(128, 4), 256, 0, stream>>>(
      e_emb16, a_emb16, nullptr, W3_16, b3, e_learn, e_learn16, BT, 128, 256, 1);
  // gx = e_learning @ gru_w_ih^T + b_ih
  gemm_cat_kernel<4><<<blocks_for(384, 4), 256, 0, stream>>>(
      e_learn16, nullptr, nullptr, gwih16, gb_ih, gx, nullptr, BT, 384, 128, 0);
  // Wall = se_data @ key_matrix^T
  gemm_cat_kernel<2><<<blocks_for(32, 2), 256, 0, stream>>>(
      se_data16, nullptr, nullptr, key16, nullptr, WallB, nullptr, BT, 32, 128, 0);

  const size_t gru_lds = (size_t)(BB * HH + BB * 384 + 384) * sizeof(float)
                       + (size_t)(BB * HH + 384 * HH) * sizeof(_Float16);
  gru_kernel<<<1, 512, gru_lds, stream>>>(gx, gwhh16, gb_hh, h0, gru_out);

  const size_t ms_lds = (size_t)(2 * KK * HH + 4 * HH) * sizeof(float)
                      + (size_t)(KK * HH + HH * HH) * sizeof(_Float16);
  memscan_kernel<<<BB, 256, ms_lds, stream>>>(WallB, se_learn, Wg, bg, Wgg, bgg, Wf, bf,
                                              m0, out1);
  output_kernel<<<(BB * (TT - 1) + 255) / 256, 256, 0, stream>>>(gru_out, out1, e_emb,
                                                                 se_data, Wp, bp, Wp1,
                                                                 bp1, pred);
}
